// myGRU_group_54571854463577
// MI455X (gfx1250) — compile-verified, matrix operand order checked by reading, not run on hardware
//
#include <hip/hip_runtime.h>
#include <hip/hip_bf16.h>

// ---------------------------------------------------------------------------
// VMLMF-group GRU for MI455X (gfx1250, wave32, WMMA + TDM).
//   B=256, T=512, N=h=256, 3h=768, L=2 layers.
// Strategy:
//   1) Densify low-rank factors: WeffT/UeffT = (Ad@Gd + Ao@Go)^T as bf16,
//      stored [768][256] (N-major) so WMMA B-fragments are contiguous.
//   2) wx GEMM [131072,256]x[256,768]: B panel DMA'd into LDS once per block
//      via tensor_load_to_lds (TDM), A tiles staged f32->bf16 vectorized,
//      compute with v_wmma_f32_16x16x32_bf16.
//   3) Persistent single-WGP scan: h kept bf16 in LDS (2x128KB double buffer,
//      fits CDNA5's 320KB/WGP), Ueff streamed from L2, gates in f32 VALU.
// ---------------------------------------------------------------------------

typedef __attribute__((ext_vector_type(16))) __bf16 v16bf;
typedef __attribute__((ext_vector_type(8)))  __bf16 v8bf;
typedef __attribute__((ext_vector_type(4)))  __bf16 v4bf;
typedef __attribute__((ext_vector_type(8)))  float  v8f;
typedef __attribute__((ext_vector_type(4)))  float  f32x4;
typedef __attribute__((ext_vector_type(4)))  unsigned int u32x4;
typedef __attribute__((ext_vector_type(8)))  int    i32x8;
typedef __attribute__((ext_vector_type(4)))  int    i32x4;

#define LAYERS 2
#define BB 256
#define TT 512
#define NN 256      // input dim == hidden dim
#define HH 256
#define G3 768      // 3*h

// ---------------------------------------------------------------------------
// Effective weight build: outT[l][j][k] = sum_r A1[l][k][r]*G1[l][r][j]
//                                       + sum_r A2[l][k][r]*G2[l][r][j]
// Stored transposed (j-major) as bf16 for contiguous WMMA B-fragment loads.
// ---------------------------------------------------------------------------
__global__ __launch_bounds__(256)
void build_eff_T(const float* __restrict__ A1, const float* __restrict__ G1, int r1,
                 const float* __restrict__ A2, const float* __restrict__ G2, int r2,
                 __bf16* __restrict__ outT) {
  int idx = blockIdx.x * blockDim.x + threadIdx.x;
  if (idx >= LAYERS * G3 * NN) return;
  int l   = idx / (G3 * NN);
  int rem = idx - l * (G3 * NN);
  int j   = rem >> 8;     // / 256
  int k   = rem & 255;
  const float* a1 = A1 + (size_t)l * NN * r1;
  const float* g1 = G1 + (size_t)l * r1 * G3;
  const float* a2 = A2 + (size_t)l * NN * r2;
  const float* g2 = G2 + (size_t)l * r2 * G3;
  float acc = 0.f;
  for (int r = 0; r < r1; ++r) acc += a1[k * r1 + r] * g1[r * G3 + j];
  for (int r = 0; r < r2; ++r) acc += a2[k * r2 + r] * g2[r * G3 + j];
  outT[(size_t)l * G3 * NN + (size_t)j * NN + k] = (__bf16)acc;
}

// ---------------------------------------------------------------------------
// wx GEMM:  wx[M=131072][768] = X[M][256] @ Weff[256][768] + bias
// Block: 256 threads (8 wave32), tile 128(M) x 64(N).
// B panel (64 N rows x 256 K, bf16, 32KB) loaded once per block by the
// Tensor Data Mover; A tiles staged per-K-step with vectorized f32->bf16.
// Each wave owns one 16-row M tile and 4 N tiles (4 f32 accumulators).
// ---------------------------------------------------------------------------
__global__ __launch_bounds__(256)
void wx_gemm(const float*  __restrict__ X,    // [M][256] fp32
             const __bf16* __restrict__ WT,   // [768][256] bf16 (transposed)
             const float*  __restrict__ bias, // [768]
             float*        __restrict__ wx) { // [M][768]
  __shared__ __bf16 As[128 * 32];    //  8 KB (one K step)
  __shared__ __bf16 Bs[64 * 256];    // 32 KB (full K panel, TDM destination)
  const int tid     = threadIdx.x;
  const int w       = tid >> 5;
  const int lane    = tid & 31;
  const int rowBase = blockIdx.x * 128;
  const int nBase   = blockIdx.y * 64;
  const int m       = lane & 15;          // fragment row/col within tile
  const int kb      = (lane >> 4) * 8;    // K-half select (ISA 16-bit A layout)

  // --- TDM: DMA the bf16 B panel [64][256] into LDS (wave 0 issues once) ---
  if (tid < 32) {
    unsigned long long gaddr =
        (unsigned long long)(uintptr_t)(WT + (size_t)nBase * NN);
    unsigned int lds_addr = (unsigned int)(uintptr_t)&Bs[0];  // low 32 = LDS off
    u32x4 g0;
    g0[0] = 1u;                                            // count=1, user D#
    g0[1] = lds_addr;                                      // lds_addr[31:0]
    g0[2] = (unsigned int)gaddr;                           // global_addr[31:0]
    g0[3] = (unsigned int)((gaddr >> 32) & 0x1FFFFFFull)   // global_addr[56:32]
          | (2u << 30);                                    // type=2 (image)
    i32x8 g1;
    g1[0] = (int)(1u << 16);          // workgroup_mask=0, data_size=1 (2B)
    g1[1] = (int)(256u << 16);        // tensor_dim0[15:0]=256 in [31:16]
    g1[2] = (int)(768u << 16);        // tensor_dim0 hi=0; tensor_dim1 lo=768
    g1[3] = (int)(256u << 16);        // tensor_dim1 hi=0; tile_dim0=256
    g1[4] = (int)64u;                 // tile_dim1=64; tile_dim2=0
    g1[5] = (int)256u;                // tensor_dim0_stride lo32 = 256
    g1[6] = 0;                        // stride0 hi; stride1 lo
    g1[7] = 0;                        // stride1 hi
    i32x4 gz4 = {0, 0, 0, 0};
    i32x8 gz8 = {0, 0, 0, 0, 0, 0, 0, 0};
    __builtin_amdgcn_tensor_load_to_lds(g0, g1, gz4, gz4, gz8, 0);
    __builtin_amdgcn_s_wait_tensorcnt(0);
  }
  __syncthreads();   // Bs ready for all waves

  v8f acc[4] = {v8f{}, v8f{}, v8f{}, v8f{}};

  for (int k8 = 0; k8 < 8; ++k8) {
    const int k0 = k8 * 32;
    // Stage A tile: vectorized fp32 global -> packed bf16 LDS
    // 128x32 f32 = 1024 float4 chunks; 4 per thread.
#pragma unroll
    for (int e = 0; e < 4; ++e) {
      int c4  = e * 256 + tid;          // float4 chunk id
      int r   = c4 >> 3;                // row (32 cols / 4 = 8 chunks per row)
      int col = (c4 & 7) * 4;
      f32x4 fv = *(const f32x4*)&X[(size_t)(rowBase + r) * NN + k0 + col];
      v4bf bv;
      bv[0] = (__bf16)fv[0]; bv[1] = (__bf16)fv[1];
      bv[2] = (__bf16)fv[2]; bv[3] = (__bf16)fv[3];
      *(v4bf*)&As[r * 32 + col] = bv;
    }
    if (k8 < 7)  // hint next A slice into cache (global_prefetch_b8)
      __builtin_prefetch(&X[(size_t)(rowBase + (tid >> 1)) * NN + k0 + 32], 0, 1);
    __syncthreads();

    // A fragment (16x32 bf16, §7.12.2 layout): two contiguous 16B LDS loads
    v16bf a;
    ((v8bf*)&a)[0] = *(const v8bf*)&As[(w * 16 + m) * 32 + kb];
    ((v8bf*)&a)[1] = *(const v8bf*)&As[(w * 16 + m) * 32 + kb + 16];
#pragma unroll
    for (int nt = 0; nt < 4; ++nt) {
      v16bf bf;
      ((v8bf*)&bf)[0] = *(const v8bf*)&Bs[(nt * 16 + m) * 256 + k0 + kb];
      ((v8bf*)&bf)[1] = *(const v8bf*)&Bs[(nt * 16 + m) * 256 + k0 + kb + 16];
      acc[nt] = __builtin_amdgcn_wmma_f32_16x16x32_bf16(
          false, a, false, bf, (short)0, acc[nt], false, false);
    }
    __syncthreads();   // protect As before next stage overwrites it
  }

  // Epilogue: D layout -> lane holds rows (lane>>4)*8 + e, col lane&15
  const int rb = (lane >> 4) * 8;
#pragma unroll
  for (int nt = 0; nt < 4; ++nt) {
    const int col = nBase + nt * 16 + m;
    const float bv = bias[col];
#pragma unroll
    for (int e = 0; e < 8; ++e)
      wx[(size_t)(rowBase + w * 16 + rb + e) * G3 + col] = acc[nt][e] + bv;
  }
}

// ---------------------------------------------------------------------------
// GRU recurrent scan, one layer. Single workgroup (32 wave32 waves) persists
// over T steps; h lives bf16 in LDS (double-buffered, 2x128KB).
// Each wave handles 8 (mtile,qtile) pairs; for each pair it accumulates the
// z/r/c gate tiles (3 WMMA accumulators) so gates fuse entirely in registers.
// ---------------------------------------------------------------------------
__global__ __launch_bounds__(1024)
void gru_scan(const float*  __restrict__ wx,  // [B*T][768] (bias folded in)
              const __bf16* __restrict__ UT,  // [768][256] bf16 (transposed)
              float*        __restrict__ y,   // [B][T][256] layer output
              float*        __restrict__ hT,  // final hidden: index b*512 + q
              int T) {
  extern __shared__ __bf16 hbuf[];            // 2 * 256*256 bf16 = 256 KB
  const int tid  = threadIdx.x;
  const int w    = tid >> 5;
  const int lane = tid & 31;
  const int m    = lane & 15;
  const int kb   = (lane >> 4) * 8;
  const int rb   = (lane >> 4) * 8;

  for (int i = tid; i < 2 * HH * HH; i += 1024) hbuf[i] = (__bf16)0.f;
  __syncthreads();

  for (int t = 0; t < T; ++t) {
    __bf16* hc  = hbuf + (t & 1) * (HH * HH);
    __bf16* hnx = hbuf + ((t & 1) ^ 1) * (HH * HH);

    for (int p = 0; p < 8; ++p) {            // 8 (mtile,qtile) pairs per wave
      const int pairIdx = w + 32 * p;
      const int mt = pairIdx >> 4;
      const int qt = pairIdx & 15;
      v8f cz = {}, cr = {}, cc = {};
#pragma unroll
      for (int k8 = 0; k8 < 8; ++k8) {       // K = 256 in steps of 32
        const __bf16* ap = hc + (mt * 16 + m) * HH + k8 * 32 + kb;
        v16bf a;
        ((v8bf*)&a)[0] = *(const v8bf*)ap;
        ((v8bf*)&a)[1] = *(const v8bf*)(ap + 16);
        {  // z gate (cols 0..255)
          const __bf16* bp = UT + (size_t)(qt * 16 + m) * HH + k8 * 32 + kb;
          v16bf bf;
          ((v8bf*)&bf)[0] = *(const v8bf*)bp;
          ((v8bf*)&bf)[1] = *(const v8bf*)(bp + 16);
          cz = __builtin_amdgcn_wmma_f32_16x16x32_bf16(false, a, false, bf,
                                                       (short)0, cz, false, false);
        }
        {  // r gate (cols 256..511)
          const __bf16* bp = UT + (size_t)(HH + qt * 16 + m) * HH + k8 * 32 + kb;
          v16bf bf;
          ((v8bf*)&bf)[0] = *(const v8bf*)bp;
          ((v8bf*)&bf)[1] = *(const v8bf*)(bp + 16);
          cr = __builtin_amdgcn_wmma_f32_16x16x32_bf16(false, a, false, bf,
                                                       (short)0, cr, false, false);
        }
        {  // c candidate (cols 512..767)
          const __bf16* bp = UT + (size_t)(2 * HH + qt * 16 + m) * HH + k8 * 32 + kb;
          v16bf bf;
          ((v8bf*)&bf)[0] = *(const v8bf*)bp;
          ((v8bf*)&bf)[1] = *(const v8bf*)(bp + 16);
          cc = __builtin_amdgcn_wmma_f32_16x16x32_bf16(false, a, false, bf,
                                                       (short)0, cc, false, false);
        }
      }
      // Fused gate math in f32; write h(t+1) into the other LDS buffer.
      const int q = qt * 16 + m;
#pragma unroll
      for (int e = 0; e < 8; ++e) {
        const int b = mt * 16 + rb + e;
        const size_t wrow = ((size_t)b * T + t) * G3;
        float z  = wx[wrow + q]          + cz[e];
        float r  = wx[wrow + HH + q]     + cr[e];
        float cg = wx[wrow + 2 * HH + q];
        float hp = (float)hc[b * HH + q];
        z  = 1.f / (1.f + __expf(-z));
        r  = 1.f / (1.f + __expf(-r));
        cg = tanhf(cg + r * cc[e]);
        float hv = z * hp + (1.f - z) * cg;
        hnx[b * HH + q] = (__bf16)hv;
        y[((size_t)b * T + t) * HH + q] = hv;
        if (t == T - 1) hT[(size_t)b * 512 + q] = hv;
      }
    }
    __syncthreads();   // all waves done with hc / done writing hnx
  }
}

// ---------------------------------------------------------------------------
// Host launcher. Inputs (setup_inputs order):
//   0:x 1:Wd 2:Wdg 3:Wo 4:Wog 5:Ud 6:Udg 7:Uo 8:Uog 9:b
// Output: [B,T,256] sequence (33,554,432 f32) then [B,512] final hiddens.
// ---------------------------------------------------------------------------
extern "C" void kernel_launch(void* const* d_in, const int* in_sizes, int n_in,
                              void* d_out, int out_size, void* d_ws, size_t ws_size,
                              hipStream_t stream) {
  const float* x   = (const float*)d_in[0];
  const float* Wd  = (const float*)d_in[1];
  const float* Wdg = (const float*)d_in[2];
  const float* Wo  = (const float*)d_in[3];
  const float* Wog = (const float*)d_in[4];
  const float* Ud  = (const float*)d_in[5];
  const float* Udg = (const float*)d_in[6];
  const float* Uo  = (const float*)d_in[7];
  const float* Uog = (const float*)d_in[8];
  const float* b   = (const float*)d_in[9];
  float* out = (float*)d_out;

  // Workspace layout (bytes):
  //   [0,786432)                      WeffT bf16 [2][768][256]
  //   [786432,1572864)                UeffT bf16 [2][768][256]
  //   [1572864, +402653184)           wx    f32  [131072][768]  (reused per layer)
  //   [+402653184, +134217728)        y0    f32  [256][512][256]
  char* ws = (char*)d_ws;
  __bf16* WeffT = (__bf16*)ws;
  __bf16* UeffT = (__bf16*)(ws + 786432);
  float*  wxbuf = (float*)(ws + 1572864);
  float*  y0    = (float*)(ws + 1572864 + 402653184ull);

  const int effN = LAYERS * G3 * NN;
  build_eff_T<<<(effN + 255) / 256, 256, 0, stream>>>(Wd, Wdg, 64, Wo, Wog, 32, WeffT);
  build_eff_T<<<(effN + 255) / 256, 256, 0, stream>>>(Ud, Udg, 64, Uo, Uog, 32, UeffT);

  const float* Xs[2] = { x, y0 };
  float*       Ys[2] = { y0, out };
  for (int l = 0; l < 2; ++l) {
    dim3 gg((BB * TT) / 128, G3 / 64);
    wx_gemm<<<gg, 256, 0, stream>>>(Xs[l], WeffT + (size_t)l * G3 * NN,
                                    b + (size_t)l * G3, wxbuf);
    gru_scan<<<1, 1024, 2 * HH * HH * (int)sizeof(__bf16), stream>>>(
        wxbuf, UeffT + (size_t)l * G3 * NN, Ys[l],
        out + (size_t)BB * TT * HH + (size_t)l * HH, TT);
  }
}
